// Attention_53712861003822
// MI455X (gfx1250) — matrix operand order repaired and checked by reading, not compiled, not run on hardware
//
#include <hip/hip_runtime.h>

typedef __attribute__((ext_vector_type(16))) _Float16 v16h;
typedef __attribute__((ext_vector_type(8)))  _Float16 v8h;
typedef __attribute__((ext_vector_type(8)))  float    v8f;

constexpr int kB = 4, kS = 1024, kD = 4096;
constexpr int kHQ = 32, kHKV = 8, kHD = 128, kGroups = 4;
constexpr int kNQKV = kHQ * kHD + 2 * kHKV * kHD;   // 6144
constexpr int kTok  = kB * kS;                      // 4096
constexpr float kScale = 0.08838834764831845f;      // 1/sqrt(128)
constexpr float kLog2e = 1.4426950408889634f;

__device__ __forceinline__ v8f zero8() {
  v8f z = {0.f, 0.f, 0.f, 0.f, 0.f, 0.f, 0.f, 0.f};
  return z;
}

// 16-bit A-matrix fragment (16x32): lane holds row m = lane%16,
// K contiguous [half*16, half*16+16).
__device__ __forceinline__ v16h load_afrag(const _Float16* rowk, int half) {
  union { v16h v; v8h h[2]; } u;
  u.h[0] = *(const v8h*)(rowk + half * 16);
  u.h[1] = *(const v8h*)(rowk + half * 16 + 8);
  return u.v;
}

__device__ __forceinline__ v8f wmma16(v16h a, v16h b, v8f c) {
  return __builtin_amdgcn_wmma_f32_16x16x32_f16(false, a, false, b, (short)0, c,
                                                false, false);
}

// ---------------- elementwise converts ----------------
__global__ void f32_to_f16_kernel(const float* __restrict__ src,
                                  _Float16* __restrict__ dst, int n) {
  int i = blockIdx.x * blockDim.x + threadIdx.x;
  if (i < n) dst[i] = (_Float16)src[i];
}

__global__ void convert_w_kernel(const float* __restrict__ Wq,
                                 const float* __restrict__ Wk,
                                 const float* __restrict__ Wv,
                                 _Float16* __restrict__ Wh) {
  int i = blockIdx.x * blockDim.x + threadIdx.x;  // < 6144*4096
  if (i >= kNQKV * kD) return;
  int row = i >> 12;        // /4096
  int colk = i & 4095;
  float v;
  if (row < 4096)       v = Wq[i];
  else if (row < 5120)  v = Wk[(size_t)(row - 4096) * kD + colk];
  else                  v = Wv[(size_t)(row - 5120) * kD + colk];
  Wh[i] = (_Float16)v;
}

// ---------------- RoPE (in place on QKVh, Q + K sections) ----------------
__global__ void rope_kernel(_Float16* __restrict__ QKVh,
                            const float* __restrict__ cosp,
                            const float* __restrict__ sinp) {
  int i = blockIdx.x * blockDim.x + threadIdx.x;  // < 4096*40*64
  if (i >= kTok * (kHQ + kHKV) * 64) return;
  int d = i & 63;
  int tmp = i >> 6;
  int head = tmp % (kHQ + kHKV);
  int t = tmp / (kHQ + kHKV);
  int s = t & (kS - 1);
  int base = (head < kHQ) ? head * kHD : kHQ * kHD + (head - kHQ) * kHD;
  size_t o = (size_t)t * kNQKV + base + d;
  float x1 = (float)QKVh[o];
  float x2 = (float)QKVh[o + 64];
  float c = cosp[(size_t)s * kHD + 64 + d];
  float sn = sinp[(size_t)s * kHD + 64 + d];
  QKVh[o]      = (_Float16)(x1 * c - x2 * sn);
  QKVh[o + 64] = (_Float16)(x2 * c + x1 * sn);
}

// ---------------- V transpose: Vt[b][hkv][d][s] ----------------
__global__ void transpose_v_kernel(const _Float16* __restrict__ QKVh,
                                   _Float16* __restrict__ Vt) {
  int i = blockIdx.x * blockDim.x + threadIdx.x;  // < 4*1024*8*128
  if (i >= kTok * kHKV * kHD) return;
  int d = i & 127;
  int tmp = i >> 7;
  int hkv = tmp & 7;
  int t = tmp >> 3;
  int b = t >> 10;
  int s = t & (kS - 1);
  Vt[((size_t)(b * kHKV + hkv) * kHD + d) * kS + s] =
      QKVh[(size_t)t * kNQKV + kHQ * kHD + kHKV * kHD + hkv * kHD + d];
}

// ---------------- WMMA GEMM: C[M,N] = A[M,K] * B[N,K]^T ----------------
// One wave computes a 32 x 64 block of C (2 M-tiles x 4 N-tiles):
// 8 WMMAs per K-step over 6 fragment loads -> ~21 FLOP/byte from L2.
template <typename TOut>
__global__ __launch_bounds__(32) void gemm_f16_kernel(
    const _Float16* __restrict__ A, const _Float16* __restrict__ Bm,
    TOut* __restrict__ C, int M, int N, int K) {
  int lane = threadIdx.x;
  int col = lane & 15, half = lane >> 4;
  int mt0 = blockIdx.y * 2;
  int nt0 = blockIdx.x * 4;
  const _Float16* Arow0 = A + (size_t)(mt0 * 16 + col) * K;
  const _Float16* Arow1 = Arow0 + (size_t)16 * K;
  v8f acc[2][4];
#pragma unroll
  for (int m = 0; m < 2; ++m)
#pragma unroll
    for (int t = 0; t < 4; ++t) acc[m][t] = zero8();
  for (int k0 = 0; k0 < K; k0 += 32) {
    v16h a0 = load_afrag(Arow0 + k0, half);
    v16h a1 = load_afrag(Arow1 + k0, half);
#pragma unroll
    for (int t = 0; t < 4; ++t) {
      const _Float16* Brow =
          Bm + (size_t)((nt0 + t) * 16 + col) * K + k0 + half * 16;
      v16h bf = *(const v16h*)Brow;
      acc[0][t] = wmma16(a0, bf, acc[0][t]);
      acc[1][t] = wmma16(a1, bf, acc[1][t]);
    }
  }
#pragma unroll
  for (int m = 0; m < 2; ++m)
#pragma unroll
    for (int t = 0; t < 4; ++t) {
      size_t base = (size_t)((mt0 + m) * 16 + half * 8) * N +
                    (size_t)(nt0 + t) * 16 + col;
#pragma unroll
      for (int r = 0; r < 8; ++r) C[base + (size_t)r * N] = (TOut)acc[m][t][r];
    }
}

// ---------------- Flash attention: one wave per (b, h, 16-query tile) ------
__global__ __launch_bounds__(32) void attn_kernel(
    const _Float16* __restrict__ QKVh, const _Float16* __restrict__ Vt,
    _Float16* __restrict__ AttnH) {
  __shared__ _Float16 Pst[16 * 32];
  int lane = threadIdx.x;
  int col = lane & 15, half = lane >> 4;
  int bh = blockIdx.x;
  int b = bh / kHQ, h = bh % kHQ, hkv = h / kGroups;
  int qt = blockIdx.y;
  const float l2s = kScale * kLog2e;

  // Q fragments (A-matrix), kept in registers for the whole key loop.
  v16h qa[4];
  {
    const _Float16* Qrow =
        QKVh + (size_t)(b * kS + qt * 16 + col) * kNQKV + h * kHD;
#pragma unroll
    for (int kk = 0; kk < 4; ++kk) qa[kk] = load_afrag(Qrow + kk * 32, half);
  }

  float m_i[8], l_i[8];
  v8f accv[8];
#pragma unroll
  for (int r = 0; r < 8; ++r) { m_i[r] = -1e30f; l_i[r] = 0.f; }
#pragma unroll
  for (int c = 0; c < 8; ++c) accv[c] = zero8();

  const size_t vhead = (size_t)(b * kHKV + hkv) * kHD;

  for (int j = 0; j < kS; j += 32) {
    // scores for 16 queries x 32 keys
    v8f sc[2];
#pragma unroll
    for (int t2 = 0; t2 < 2; ++t2) {
      sc[t2] = zero8();
      const _Float16* Kb = QKVh + (size_t)(b * kS + j + t2 * 16 + col) * kNQKV +
                           kHQ * kHD + hkv * kHD + half * 16;
#pragma unroll
      for (int kk = 0; kk < 4; ++kk) {
        v16h bf = *(const v16h*)(Kb + kk * 32);
        sc[t2] = wmma16(qa[kk], bf, sc[t2]);
      }
    }
    // online softmax: rows m = half*8 + r live in the 16-lane group of `half`
    float p0[8], p1[8], alpha[8];
#pragma unroll
    for (int r = 0; r < 8; ++r) {
      float z0 = sc[0][r] * l2s, z1 = sc[1][r] * l2s;
      float cand = fmaxf(z0, z1);
      cand = fmaxf(cand, __shfl_xor(cand, 1));
      cand = fmaxf(cand, __shfl_xor(cand, 2));
      cand = fmaxf(cand, __shfl_xor(cand, 4));
      cand = fmaxf(cand, __shfl_xor(cand, 8));
      float mn = fmaxf(m_i[r], cand);
      float a = exp2f(m_i[r] - mn);
      float e0 = exp2f(z0 - mn), e1 = exp2f(z1 - mn);
      float rs = e0 + e1;
      rs += __shfl_xor(rs, 1);
      rs += __shfl_xor(rs, 2);
      rs += __shfl_xor(rs, 4);
      rs += __shfl_xor(rs, 8);
      l_i[r] = l_i[r] * a + rs;
      m_i[r] = mn;
      alpha[r] = a; p0[r] = e0; p1[r] = e1;
    }
#pragma unroll
    for (int c = 0; c < 8; ++c)
#pragma unroll
      for (int r = 0; r < 8; ++r) accv[c][r] *= alpha[r];

    // stage exp'd P to LDS row-major [16 queries][32 keys], re-fragment as A
#pragma unroll
    for (int r = 0; r < 8; ++r) {
      Pst[(half * 8 + r) * 32 + col]      = (_Float16)p0[r];
      Pst[(half * 8 + r) * 32 + 16 + col] = (_Float16)p1[r];
    }
    __asm__ volatile("" ::: "memory");  // keep LDS store->load order (DS in-order per wave)
    v16h pa = load_afrag(&Pst[col * 32], half);

    // P @ V : B-fragment from transposed V, contiguous in keys
#pragma unroll
    for (int c = 0; c < 8; ++c) {
      v16h vb = *(const v16h*)(Vt + (vhead + c * 16 + col) * kS + j + half * 16);
      accv[c] = wmma16(pa, vb, accv[c]);
    }
  }

  // normalize and write f16 attention output [token][HQ*HD]
  float inv[8];
#pragma unroll
  for (int r = 0; r < 8; ++r) inv[r] = 1.0f / l_i[r];
  size_t tok0 = (size_t)(b * kS + qt * 16 + half * 8);
#pragma unroll
  for (int c = 0; c < 8; ++c)
#pragma unroll
    for (int r = 0; r < 8; ++r)
      AttnH[(tok0 + r) * (size_t)(kHQ * kHD) + h * kHD + c * 16 + col] =
          (_Float16)(accv[c][r] * inv[r]);
}

extern "C" void kernel_launch(void* const* d_in, const int* in_sizes, int n_in,
                              void* d_out, int out_size, void* d_ws, size_t ws_size,
                              hipStream_t stream) {
  const float* X    = (const float*)d_in[0];  // (B,S,D)
  const float* cosp = (const float*)d_in[1];  // (1,S,HD)
  const float* sinp = (const float*)d_in[2];
  const float* Wq   = (const float*)d_in[3];
  const float* Wk   = (const float*)d_in[4];
  const float* Wv   = (const float*)d_in[5];
  const float* Wo   = (const float*)d_in[6];
  float* out = (float*)d_out;

  // workspace layout (bytes), buffers reused across phases:
  //  [0,   50331648)  QKVh  f16 (4096 x 6144)
  //  [.. , +8388608)  Vt    f16 (4*8*128*1024)
  //  [.. , +33554432) Xh (phase 1) / AttnH (phase 3)  f16 (4096 x 4096)
  //  [.. , +50331648) Wh=[Wq|Wk|Wv] (phase 1) / Woh (phase 3)  f16
  char* w = (char*)d_ws;
  _Float16* QKVh = (_Float16*)w;
  _Float16* Vt   = (_Float16*)(w + 50331648ull);
  _Float16* XA   = (_Float16*)(w + 50331648ull + 8388608ull);
  _Float16* WW   = (_Float16*)(w + 50331648ull + 8388608ull + 33554432ull);

  const int nX = kTok * kD;            // 16,777,216
  const int nW = kNQKV * kD;           // 25,165,824
  const int nRope = kTok * (kHQ + kHKV) * 64;   // 10,485,760
  const int nVt = kTok * kHKV * kHD;   // 4,194,304

  // 1) converts
  f32_to_f16_kernel<<<nX / 256, 256, 0, stream>>>(X, XA, nX);
  convert_w_kernel<<<nW / 256, 256, 0, stream>>>(Wq, Wk, Wv, WW);
  // 2) fused QKV projection (f16 out), 32x64 per wave
  gemm_f16_kernel<_Float16><<<dim3(kNQKV / 64, kTok / 32), 32, 0, stream>>>(
      XA, WW, QKVh, kTok, kNQKV, kD);
  // 3) RoPE in place on Q/K sections
  rope_kernel<<<nRope / 256, 256, 0, stream>>>(QKVh, cosp, sinp);
  // 4) V transpose for P@V fragments
  transpose_v_kernel<<<nVt / 256, 256, 0, stream>>>(QKVh, Vt);
  // 5) Wo -> f16 (reuses the Wq|Wk|Wv slot; GEMM1 already consumed it)
  f32_to_f16_kernel<<<nX / 256, 256, 0, stream>>>(Wo, WW, nX);
  // 6) flash attention (AttnH overwrites Xh slot)
  attn_kernel<<<dim3(kB * kHQ, kS / 16), 32, 0, stream>>>(QKVh, Vt, XA);
  // 7) output projection, fp32 out
  gemm_f16_kernel<float><<<dim3(kD / 64, kTok / 32), 32, 0, stream>>>(
      XA, WW, out, kTok, kD, kD);
}
